// PointNet2Multiview2Msg_7954279432402
// MI455X (gfx1250) — compile-verified, matrix operand order checked by reading, not run on hardware
//
#include <hip/hip_runtime.h>
#include <hip/hip_bf16.h>

#define BB 4
#define NN 8192
#define RND32(x) (((x) + 31) & ~31)

typedef unsigned short bf16_t;
typedef __attribute__((ext_vector_type(16))) __bf16 v16bf;
typedef __attribute__((ext_vector_type(8)))  float  v8f;

__device__ __forceinline__ bf16_t f2bf(float f) {
  unsigned int u = __float_as_uint(f);
  u += 0x7FFFu + ((u >> 16) & 1u);          // round-to-nearest-even
  return (bf16_t)(u >> 16);
}
__device__ __forceinline__ float bf2f(bf16_t h) {
  return __uint_as_float(((unsigned int)h) << 16);
}

__device__ __forceinline__ void wait_asynccnt0() {
#if __has_builtin(__builtin_amdgcn_s_wait_asynccnt)
  __builtin_amdgcn_s_wait_asynccnt(0);
#else
  asm volatile("s_wait_asynccnt 0x0" ::: "memory");
#endif
}

// async memory->LDS copy of 16 bytes per lane (ASYNCcnt-tracked, no VGPR data)
__device__ __forceinline__ void async_b128(unsigned lds_off, unsigned long long gaddr) {
  asm volatile("global_load_async_to_lds_b128 %0, %1, off"
               :: "v"(lds_off), "v"(gaddr) : "memory");
}

// ---------------------------------------------------------------------------
// WMMA GEMM:  Y[M,Nc] = act( (X[M,K] . W[Nc,K]^T) * g + b )
// X: bf16 [M, ldx], W: bf16 padded [Ncp, ldw]; K multiple of 32; M multiple
// of 64; block = 128 thr = 4 waves; macro-tile 64(M) x 32(N); 2 WMMA / chunk.
// Staging via GLOBAL_LOAD_ASYNC_TO_LDS_B128 (pure aligned bf16 copy).
// ---------------------------------------------------------------------------
__global__ __launch_bounds__(128) void mlp_wmma_kernel(
    const bf16_t* __restrict__ X, int ldx,
    const bf16_t* __restrict__ W, int ldw,
    const float* __restrict__ G, const float* __restrict__ Bv,
    void* __restrict__ Yp, int ldy,
    long long M, int K, int Nc, int relu, int ybf16)
{
  __shared__ bf16_t lx[64][40];   // 40: keeps 16B alignment, staggers banks
  __shared__ bf16_t lw[32][40];
  const int tid  = threadIdx.x;
  const int wave = tid >> 5;
  const int lane = tid & 31;
  const int hi   = lane >> 4;
  const int lmod = lane & 15;
  const long long m0 = (long long)blockIdx.x * 64;
  const int n0 = blockIdx.y * 32;

  // staging coordinates (b128 = 8 bf16 per lane)
  const int xr0 = tid >> 2, xq = (tid & 3) * 8;        // rows 0..31
  const int xr1 = xr0 + 32;                            // rows 32..63
  const int wr  = tid >> 2, wq = (tid & 3) * 8;        // 32 rows of W

  // LDS byte offsets (low 32 bits of generic shared address == LDS offset)
  const unsigned ldsx0 = (unsigned)(size_t)&lx[xr0][xq];
  const unsigned ldsx1 = (unsigned)(size_t)&lx[xr1][xq];
  const unsigned ldsw0 = (unsigned)(size_t)&lw[wr][wq];
  const unsigned long long gx0 = (unsigned long long)(size_t)(X + (m0 + xr0) * (long long)ldx + xq);
  const unsigned long long gx1 = (unsigned long long)(size_t)(X + (m0 + xr1) * (long long)ldx + xq);
  const unsigned long long gw0 = (unsigned long long)(size_t)(W + ((long long)(n0 + wr)) * ldw + wq);

  v8f acc0 = {}, acc1 = {};

  for (int k0 = 0; k0 < K; k0 += 32) {
    const unsigned long long koff = (unsigned long long)k0 * 2ull;
    async_b128(ldsx0, gx0 + koff);
    async_b128(ldsx1, gx1 + koff);
    async_b128(ldsw0, gw0 + koff);
    if (k0 + 32 < K)
      __builtin_prefetch(X + (m0 + (tid >> 1)) * (long long)ldx + k0 + 32, 0, 1);
    wait_asynccnt0();          // own async LDS writes landed
    __syncthreads();           // everyone's writes visible

    const int mrow = wave * 16 + lmod;
    union { v16bf v; unsigned int u[8]; } a, b0, b1;
#pragma unroll
    for (int q = 0; q < 8; ++q) {
      int kk = (q < 4) ? (2 * q + 8 * hi) : (8 + 2 * q + 8 * hi);
      a.u[q]  = *(const unsigned int*)&lx[mrow][kk];
      b0.u[q] = *(const unsigned int*)&lw[lmod][kk];
      b1.u[q] = *(const unsigned int*)&lw[16 + lmod][kk];
    }
    acc0 = __builtin_amdgcn_wmma_f32_16x16x32_bf16(false, a.v, false, b0.v,
                                                   (short)0, acc0, false, false);
    acc1 = __builtin_amdgcn_wmma_f32_16x16x32_bf16(false, a.v, false, b1.v,
                                                   (short)0, acc1, false, false);
    __syncthreads();           // protect LDS against next chunk's overwrite
  }

  // epilogue: scale * g + b, relu, store (M % 64 == 0 -> no row guard)
  v8f accs[2] = { acc0, acc1 };
#pragma unroll
  for (int t = 0; t < 2; ++t) {
    const int col = n0 + t * 16 + lmod;
    if (col < Nc) {
      const float gsc = G  ? G[col]  : 1.0f;
      const float bia = Bv ? Bv[col] : 0.0f;
      union { v8f v; float f[8]; } r_; r_.v = accs[t];
#pragma unroll
      for (int r = 0; r < 8; ++r) {
        long long row = m0 + wave * 16 + r + 8 * hi;
        float val = r_.f[r] * gsc + bia;
        if (relu) val = fmaxf(val, 0.0f);
        if (ybf16) ((bf16_t*)Yp)[row * (long long)ldy + col] = f2bf(val);
        else       ((float*) Yp)[row * (long long)ldy + col] = val;
      }
    }
  }
}

// f32 weight [Nc,K] -> bf16 zero-padded [Ncp,Kp]
__global__ void wpad_kernel(const float* __restrict__ w, bf16_t* __restrict__ o,
                            int Nc, int K, int Ncp, int Kp) {
  long long t = (long long)blockIdx.x * blockDim.x + threadIdx.x;
  long long total = (long long)Ncp * Kp;
  if (t >= total) return;
  int c = (int)(t % Kp);
  long long r = t / Kp;
  float v = (r < Nc && c < K) ? w[r * (long long)K + c] : 0.0f;
  o[t] = f2bf(v);
}

// [B,3,N] f32 -> [B,N,3] f32
__global__ void transpose_xyz(const float* __restrict__ in, float* __restrict__ out, int N) {
  long long t = (long long)blockIdx.x * blockDim.x + threadIdx.x;
  long long total = (long long)BB * 3 * N;
  if (t >= total) return;
  int c = (int)(t % 3);
  long long bn = t / 3;
  int n = (int)(bn % N);
  int b = (int)(bn / N);
  out[t] = in[((long long)b * 3 + c) * N + n];
}

// [B,C,N] f32 -> [B,N,C] bf16
__global__ void transpose_feat(const float* __restrict__ in, bf16_t* __restrict__ out,
                               int C, int N) {
  long long t = (long long)blockIdx.x * blockDim.x + threadIdx.x;
  long long total = (long long)BB * C * N;
  if (t >= total) return;
  int c = (int)(t % C);
  long long bn = t / C;
  int n = (int)(bn % N);
  int b = (int)(bn / N);
  out[t] = f2bf(in[((long long)b * C + c) * N + n]);
}

// Farthest point sampling; one block per batch, iterative argmax reduction.
__global__ void fps_kernel(const float* __restrict__ xyz, int N, int npoint,
                           int* __restrict__ out_idx, float* __restrict__ dist) {
  const int b = blockIdx.x;
  const float* p = xyz + (long long)b * N * 3;
  float* d = dist + (long long)b * N;
  __shared__ float sval[256];
  __shared__ int   sidx[256];
  __shared__ int   sfar;
  if (threadIdx.x == 0) sfar = 0;
  for (int i = threadIdx.x; i < N; i += 256) d[i] = 1e10f;
  __syncthreads();
  for (int it = 0; it < npoint; ++it) {
    const int far = sfar;
    if (threadIdx.x == 0) out_idx[b * npoint + it] = far;
    const float cx = p[far*3+0], cy = p[far*3+1], cz = p[far*3+2];
    float best = -1.0f; int bi = 0;
    for (int i = threadIdx.x; i < N; i += 256) {
      float dx = p[i*3]-cx, dy = p[i*3+1]-cy, dz = p[i*3+2]-cz;
      float dd = dx*dx + dy*dy + dz*dz;
      float nd = fminf(d[i], dd);
      d[i] = nd;
      if (nd > best) { best = nd; bi = i; }
    }
    sval[threadIdx.x] = best; sidx[threadIdx.x] = bi;
    __syncthreads();
    for (int s = 128; s > 0; s >>= 1) {
      if (threadIdx.x < s && sval[threadIdx.x + s] > sval[threadIdx.x]) {
        sval[threadIdx.x] = sval[threadIdx.x + s];
        sidx[threadIdx.x] = sidx[threadIdx.x + s];
      }
      __syncthreads();
    }
    if (threadIdx.x == 0) sfar = sidx[0];
    __syncthreads();
  }
}

__global__ void gather_xyz_kernel(const float* __restrict__ xyz, const int* __restrict__ idx,
                                  float* __restrict__ out, int N, int M) {
  long long t = (long long)blockIdx.x * blockDim.x + threadIdx.x;
  long long total = (long long)BB * M * 3;
  if (t >= total) return;
  int d = (int)(t % 3);
  long long bm = t / 3;
  int m = (int)(bm % M);
  int b = (int)(bm / M);
  int s = idx[(long long)b * M + m];
  if (s < 0 || s >= N) s = 0;
  out[t] = xyz[((long long)b * N + s) * 3 + d];
}

// first-K-in-radius, ascending index order (matches sorted-idx reference)
__global__ void ball_query_kernel(const float* __restrict__ xyz, const float* __restrict__ nxyz,
                                  int N, int M, float r2, int K, int* __restrict__ out) {
  int t = blockIdx.x * blockDim.x + threadIdx.x;
  if (t >= BB * M) return;
  int b = t / M;
  const float* q = nxyz + (long long)t * 3;
  const float qx = q[0], qy = q[1], qz = q[2];
  const float* p = xyz + (long long)b * N * 3;
  int* o = out + (long long)t * K;
  int cnt = 0, first = 0; bool found = false;
  for (int i = 0; i < N && cnt < K; ++i) {
    float dx = p[i*3]-qx, dy = p[i*3+1]-qy, dz = p[i*3+2]-qz;
    float dd = dx*dx + dy*dy + dz*dz;
    if (dd <= r2) {
      if (!found) { found = true; first = i; }
      o[cnt++] = i;
    }
  }
  for (int j = cnt; j < K; ++j) o[j] = first;
}

// grouped rows [B*M*K, ldo(bf16)] = [gathered feat | xyz - center | 0-pad]
__global__ void group_kernel(const bf16_t* __restrict__ feat, int ldf,
                             const float* __restrict__ xyz,
                             const float* __restrict__ nxyz, const int* __restrict__ idx,
                             bf16_t* __restrict__ out, int ldo,
                             int N, int M, int K, int Cf) {
  long long t = (long long)blockIdx.x * blockDim.x + threadIdx.x;
  long long total = (long long)BB * M * K * ldo;
  if (t >= total) return;
  int c = (int)(t % ldo);
  long long row = t / ldo;
  int k = (int)(row % K);
  long long bm = row / K;
  int m = (int)(bm % M);
  int b = (int)(bm / M);
  int s = idx[((long long)b * M + m) * K + k];
  if (s < 0 || s >= N) s = 0;
  bf16_t v;
  if (c < Cf) v = feat[((long long)b * N + s) * ldf + c];
  else if (c < Cf + 3) {
    int d = c - Cf;
    v = f2bf(xyz[((long long)b * N + s) * 3 + d] - nxyz[((long long)b * M + m) * 3 + d]);
  } else v = 0;
  out[t] = v;
}

__global__ void maxpool_kernel(const bf16_t* __restrict__ in, int ldin,
                               bf16_t* __restrict__ out, int ldout, int off,
                               long long rows, int K, int C) {
  long long t = (long long)blockIdx.x * blockDim.x + threadIdx.x;
  long long total = rows * C;
  if (t >= total) return;
  long long r = t / C; int c = (int)(t % C);
  const bf16_t* p = in + (r * K) * (long long)ldin + c;
  float best = -3.4e38f;
  for (int k = 0; k < K; ++k) best = fmaxf(best, bf2f(p[(long long)k * ldin]));
  out[r * (long long)ldout + off + c] = f2bf(best);
}

__global__ void copy_cols_kernel(const bf16_t* __restrict__ src, int lsrc,
                                 bf16_t* __restrict__ dst, int ldst, int off,
                                 long long rows, int Cs) {
  long long t = (long long)blockIdx.x * blockDim.x + threadIdx.x;
  long long total = rows * Cs;
  if (t >= total) return;
  long long r = t / Cs; int c = (int)(t % Cs);
  dst[r * (long long)ldst + off + c] = src[r * (long long)lsrc + c];
}

// 3-NN inverse-distance interpolation -> dst[:, off:off+C2]
__global__ void three_interp_kernel(const float* __restrict__ xyz1, const float* __restrict__ xyz2,
                                    const bf16_t* __restrict__ pts2, int ld2,
                                    bf16_t* __restrict__ dst, int ldd, int off,
                                    int N1, int N2, int C2) {
  int t = blockIdx.x * blockDim.x + threadIdx.x;
  if (t >= BB * N1) return;
  int b = t / N1;
  const float* q = xyz1 + (long long)t * 3;
  float qx = q[0], qy = q[1], qz = q[2];
  const float* p = xyz2 + (long long)b * N2 * 3;
  float d0 = 3.4e38f, d1 = 3.4e38f, d2v = 3.4e38f;
  int i0 = 0, i1 = 0, i2 = 0;
  for (int s = 0; s < N2; ++s) {
    float dx = p[s*3]-qx, dy = p[s*3+1]-qy, dz = p[s*3+2]-qz;
    float dd = dx*dx + dy*dy + dz*dz;
    if (dd < d0)      { d2v=d1; i2=i1; d1=d0; i1=i0; d0=dd; i0=s; }
    else if (dd < d1) { d2v=d1; i2=i1; d1=dd; i1=s; }
    else if (dd < d2v){ d2v=dd; i2=s; }
  }
  float w0 = 1.f/(d0+1e-8f), w1 = 1.f/(d1+1e-8f), w2 = 1.f/(d2v+1e-8f);
  float ws = w0 + w1 + w2; w0 /= ws; w1 /= ws; w2 /= ws;
  const bf16_t* r0 = pts2 + ((long long)b * N2 + i0) * ld2;
  const bf16_t* r1 = pts2 + ((long long)b * N2 + i1) * ld2;
  const bf16_t* r2 = pts2 + ((long long)b * N2 + i2) * ld2;
  bf16_t* o = dst + (long long)t * ldd + off;
  for (int c = 0; c < C2; ++c)
    o[c] = f2bf(w0*bf2f(r0[c]) + w1*bf2f(r1[c]) + w2*bf2f(r2[c]));
}

// ---------------------------------------------------------------------------
// host-side orchestration
// ---------------------------------------------------------------------------
static inline unsigned gr(long long tot) { return (unsigned)((tot + 255) / 256); }

struct Bump {
  char* base; size_t off;
  float*  f(size_t n){ float*  p=(float*) (base+off); off += ((n*4+255)&~(size_t)255); return p; }
  int*    i(size_t n){ int*    p=(int*)   (base+off); off += ((n*4+255)&~(size_t)255); return p; }
  bf16_t* h(size_t n){ bf16_t* p=(bf16_t*)(base+off); off += ((n*2+255)&~(size_t)255); return p; }
};

struct PC {
  void* const* in; int i; int n;
  const float* next(){ int j = (i < n) ? i : (n - 1); ++i; return (const float*)in[j]; }
};

// Runs an MLP chain; X bf16 [M, RND32(cin)]. M must be a multiple of 64.
static bf16_t* run_chain(const bf16_t* X, long long M, int cin, const int* widths, int nl,
                         PC& pc, Bump& bump, hipStream_t stream,
                         void* finalDst, int finalLd, int finalBf16) {
  const bf16_t* cur = X;
  int Kp = RND32(cin);
  bf16_t* outp = nullptr;
  for (int l = 0; l < nl; ++l) {
    const float* w = pc.next(); const float* g = pc.next(); const float* bb = pc.next();
    int co  = widths[l];
    int ncp = RND32(co);
    bf16_t* wpad = bump.h((size_t)ncp * Kp);
    wpad_kernel<<<gr((long long)ncp * Kp), 256, 0, stream>>>(w, wpad, co, cin, ncp, Kp);
    bool last = (l == nl - 1);
    void* dst; int ldy; int yb;
    if (last && finalDst) { dst = finalDst; ldy = finalLd; yb = finalBf16; outp = (bf16_t*)finalDst; }
    else { bf16_t* d = bump.h((size_t)M * ncp); dst = d; ldy = ncp; yb = 1; outp = d; }
    dim3 grid((unsigned)(M / 64), (unsigned)(ncp / 32));
    mlp_wmma_kernel<<<grid, 128, 0, stream>>>(cur, Kp, wpad, Kp, g, bb, dst, ldy, M, Kp, co, 1, yb);
    cur = (const bf16_t*)dst; cin = co; Kp = ncp;
  }
  return outp;
}

static void run_sa(const float* xyz, int Nin, const bf16_t* feat, int ldf, int Cf,
                   int npoint, const float r2[2], const int ns[2],
                   const int w0[3], const int w1[3],
                   PC& pc, Bump& bump, size_t tmark,
                   float* new_xyz, bf16_t* pts_out, int outStride, int outOff,
                   hipStream_t stream) {
  bump.off = tmark;
  float* dist = bump.f((size_t)BB * Nin);
  int*   fidx = bump.i((size_t)BB * npoint);
  fps_kernel<<<BB, 256, 0, stream>>>(xyz, Nin, npoint, fidx, dist);
  gather_xyz_kernel<<<gr((long long)BB*npoint*3), 256, 0, stream>>>(xyz, fidx, new_xyz, Nin, npoint);

  const int* Ws[2] = { w0, w1 };
  int off = outOff;
  for (int s = 0; s < 2; ++s) {
    size_t mark2 = bump.off;
    int K = ns[s];
    int cin = Cf + 3;
    int ldo = RND32(cin);
    long long rows = (long long)BB * npoint * K;      // multiple of 64
    int* bidx = bump.i((size_t)rows);
    ball_query_kernel<<<gr((long long)BB*npoint), 256, 0, stream>>>(xyz, new_xyz, Nin, npoint, r2[s], K, bidx);
    bf16_t* grouped = bump.h((size_t)rows * ldo);
    group_kernel<<<gr(rows * ldo), 256, 0, stream>>>(feat, ldf, xyz, new_xyz, bidx,
                                                     grouped, ldo, Nin, npoint, K, Cf);
    bf16_t* last = run_chain(grouped, rows, cin, Ws[s], 3, pc, bump, stream, nullptr, 0, 1);
    maxpool_kernel<<<gr((long long)BB*npoint*Ws[s][2]), 256, 0, stream>>>(
        last, RND32(Ws[s][2]), pts_out, outStride, off, (long long)BB*npoint, K, Ws[s][2]);
    off += Ws[s][2];
    bump.off = mark2;
  }
}

static void run_fp(const float* xyz1, int N1, const float* xyz2, int N2,
                   const bf16_t* pts1, int ld1, int C1,
                   const bf16_t* pts2, int ld2, int C2,
                   const int* widths, int nl, PC& pc, Bump& bump, size_t tmark,
                   bf16_t* out, hipStream_t stream) {
  bump.off = tmark;
  int cin = C1 + C2;                 // all FP cins are multiples of 32
  long long rows = (long long)BB * N1;
  bf16_t* xb = bump.h((size_t)rows * cin);
  if (pts1) copy_cols_kernel<<<gr(rows * C1), 256, 0, stream>>>(pts1, ld1, xb, cin, 0, rows, C1);
  three_interp_kernel<<<gr((long long)BB*N1), 256, 0, stream>>>(xyz1, xyz2, pts2, ld2,
                                                                xb, cin, C1, N1, N2, C2);
  run_chain(xb, rows, cin, widths, nl, pc, bump, stream, out, widths[nl-1], 1);
}

extern "C" void kernel_launch(void* const* d_in, const int* in_sizes, int n_in,
                              void* d_out, int out_size, void* d_ws, size_t ws_size,
                              hipStream_t stream) {
  (void)in_sizes; (void)out_size; (void)ws_size;
  PC pc{ d_in, 0, n_in };
  const float* xyz = pc.next();   // [B,3,N]
  const float* img = pc.next();   // [B,128,N]
  Bump bump{ (char*)d_ws, 0 };

  // persistent buffers
  float*  xyz_t    = bump.f((size_t)BB*NN*3);
  bf16_t* feat_t   = bump.h((size_t)BB*NN*128);
  float*  l1_xyz   = bump.f((size_t)BB*1024*3);
  bf16_t* l1_pts   = bump.h((size_t)BB*1024*96);
  float*  l1_xyz_f = bump.f((size_t)BB*1024*3);
  bf16_t* l1_pts_f = bump.h((size_t)BB*1024*96);
  float*  l2_xyz   = bump.f((size_t)BB*256*3);
  float*  l2_xyz_f = bump.f((size_t)BB*256*3);
  bf16_t* l2_pts   = bump.h((size_t)BB*256*512);   // geo(256) | feat(256)
  float*  l3_xyz   = bump.f((size_t)BB*64*3);
  bf16_t* l3_pts   = bump.h((size_t)BB*64*512);
  float*  l4_xyz   = bump.f((size_t)BB*16*3);
  bf16_t* l4_pts   = bump.h((size_t)BB*16*1024);
  bf16_t* l3_pts2  = bump.h((size_t)BB*64*512);
  bf16_t* l2_pts2  = bump.h((size_t)BB*256*512);
  bf16_t* l1_pts2  = bump.h((size_t)BB*1024*256);
  bf16_t* l0_pts   = bump.h((size_t)BB*NN*128);
  bf16_t* xhead    = bump.h((size_t)BB*NN*128);
  const size_t tmark = bump.off;

  transpose_xyz <<<gr((long long)BB*3*NN),  256, 0, stream>>>(xyz, xyz_t, NN);
  transpose_feat<<<gr((long long)BB*128*NN),256, 0, stream>>>(img, feat_t, 128, NN);

  const float r1sq[2] = {0.05f*0.05f, 0.1f*0.1f};
  const float r2sq[2] = {0.1f*0.1f,   0.2f*0.2f};
  const float r3sq[2] = {0.2f*0.2f,   0.4f*0.4f};
  const float r4sq[2] = {0.4f*0.4f,   0.8f*0.8f};
  const int ns12[2] = {16, 32};
  const int sa1w0[3] = {16,16,32},    sa1w1[3] = {32,32,64};
  const int sa2w0[3] = {64,64,128},   sa2w1[3] = {64,96,128};
  const int sa3w0[3] = {128,196,256}, sa3w1[3] = {128,196,256};
  const int sa4w0[3] = {256,256,512}, sa4w1[3] = {256,384,512};

  run_sa(xyz_t,   NN,   nullptr,  0,   0,   1024, r1sq, ns12, sa1w0, sa1w1, pc, bump, tmark, l1_xyz,   l1_pts, 96,  0,   stream);
  run_sa(l1_xyz,  1024, l1_pts,   96,  96,  256,  r2sq, ns12, sa2w0, sa2w1, pc, bump, tmark, l2_xyz,   l2_pts, 512, 0,   stream);
  run_sa(xyz_t,   NN,   feat_t,   128, 128, 1024, r1sq, ns12, sa1w0, sa1w1, pc, bump, tmark, l1_xyz_f, l1_pts_f, 96, 0,  stream);
  run_sa(l1_xyz_f,1024, l1_pts_f, 96,  96,  256,  r2sq, ns12, sa2w0, sa2w1, pc, bump, tmark, l2_xyz_f, l2_pts, 512, 256, stream);
  run_sa(l2_xyz,  256,  l2_pts,   512, 512, 64,   r3sq, ns12, sa3w0, sa3w1, pc, bump, tmark, l3_xyz,   l3_pts, 512, 0,   stream);
  run_sa(l3_xyz,  64,   l3_pts,   512, 512, 16,   r4sq, ns12, sa4w0, sa4w1, pc, bump, tmark, l4_xyz,   l4_pts, 1024, 0,  stream);

  const int fp4w[2] = {512,512}, fp3w[2] = {512,512}, fp2w[2] = {256,256}, fp1w[3] = {128,128,128};
  run_fp(l3_xyz, 64,   l4_xyz, 16,  l3_pts, 512, 512, l4_pts,  1024, 1024, fp4w, 2, pc, bump, tmark, l3_pts2, stream);
  run_fp(l2_xyz, 256,  l3_xyz, 64,  l2_pts, 512, 512, l3_pts2, 512,  512,  fp3w, 2, pc, bump, tmark, l2_pts2, stream);
  run_fp(l1_xyz, 1024, l2_xyz, 256, l1_pts, 96,  96,  l2_pts2, 512,  512,  fp2w, 2, pc, bump, tmark, l1_pts2, stream);
  run_fp(xyz_t,  NN,   l1_xyz, 1024,nullptr,0,   0,   l1_pts2, 256,  256,  fp1w, 3, pc, bump, tmark, l0_pts,  stream);

  // head1 (relu) then head2 (plain linear + bias, f32 out)
  bump.off = tmark;
  {
    const int hw[1] = {128};
    run_chain(l0_pts, (long long)BB*NN, 128, hw, 1, pc, bump, stream, xhead, 128, 1);
  }
  {
    const float* w = pc.next(); const float* bb = pc.next();
    bf16_t* wpad = bump.h((size_t)32 * 128);
    wpad_kernel<<<gr((long long)32*128), 256, 0, stream>>>(w, wpad, 21, 128, 32, 128);
    dim3 grid((unsigned)(((long long)BB*NN) / 64), 1);
    mlp_wmma_kernel<<<grid, 128, 0, stream>>>(xhead, 128, wpad, 128, nullptr, bb,
                                              d_out, 21, (long long)BB*NN, 128, 21, 0, 0);
  }
}
